// MHBA_5557687681058
// MI455X (gfx1250) — matrix operand order repaired and checked by reading, not compiled, not run on hardware
//
#include <hip/hip_runtime.h>
#include <hip/hip_bf16.h>

// MI455X / gfx1250 implementation.
// Pipeline (all bf16 matrix math via v_wmma_f32_16x16x32_bf16, f32 accum):
//   K0 zero BN stats
//   K1 grouped conv + residual + exact GELU -> q (bf16), per-channel sum/sumsq
//   K2 BN affine finalize: a_e, c_e
//   K3 softmax over E + GELU -> k (bf16)
//   K3b ksum[f] = sum_d k[f,d]  (needed because BN is folded into logits)
//   K4 v = gelu(x @ lin_w^T + b), stored transposed (d-major) as bf16
//   K5 fused: S = (a_e*(Q K^T) + c_e*ksum_f)/sqrt(E); row softmax; write attn
//      (f32, an output); P stays in LDS; out = P V^T via WMMA, write f32.

typedef __attribute__((ext_vector_type(16))) __bf16 v16bf;
typedef __attribute__((ext_vector_type(8)))  float  v8f;

#define NB   128      // b' = 8 batches * 16 heads
#define ED   512      // E
#define DD   128      // D
#define BN_EPS 1e-5f
#define ATT_SCALE 0.04419417382415922f   // 1/sqrt(512)

__device__ __forceinline__ float gelu_exact(float x) {
    return 0.5f * x * (1.0f + erff(x * 0.7071067811865475f));
}

__device__ __forceinline__ v8f wmma_bf16(v16bf a, v16bf b, v8f c) {
    return __builtin_amdgcn_wmma_f32_16x16x32_bf16(
        false, a, false, b, (short)0, c, false, false);
}

// ---------------- K0: zero BN stat accumulators ----------------
__global__ void init_zero_kernel(float* p, int n) {
    int i = blockIdx.x * blockDim.x + threadIdx.x;
    if (i < n) p[i] = 0.0f;
}

// ---------------- K1: grouped conv + residual + GELU + stats ----------------
// grid (128 groups, 128 batches), block 128 (= D)
__global__ __launch_bounds__(128) void conv_gelu_stats_kernel(
        const float* __restrict__ x, const float* __restrict__ conv_w,
        const float* __restrict__ conv_b, __bf16* __restrict__ qbf,
        float* __restrict__ sum, float* __restrict__ sumsq)
{
    const int g = blockIdx.x;     // channel group (4 channels)
    const int b = blockIdx.y;
    const int d = threadIdx.x;
    __shared__ float xs[4][130];  // halo (zero padding) at both ends
    __shared__ float wsm[48];
    __shared__ float s1[4], s2[4];
    if (d < 48) wsm[d] = conv_w[g * 48 + d];
    if (d < 4)  { s1[d] = 0.f; s2[d] = 0.f; }
    #pragma unroll
    for (int i = 0; i < 4; ++i)
        xs[i][d + 1] = x[((size_t)(b * ED) + g * 4 + i) * DD + d];
    if (d == 0) {
        #pragma unroll
        for (int i = 0; i < 4; ++i) { xs[i][0] = 0.f; xs[i][129] = 0.f; }
    }
    __syncthreads();
    #pragma unroll
    for (int ol = 0; ol < 4; ++ol) {
        int o = g * 4 + ol;
        float acc = conv_b[o];
        #pragma unroll
        for (int i = 0; i < 4; ++i)
            #pragma unroll
            for (int t = 0; t < 3; ++t)
                acc += xs[i][d + t] * wsm[ol * 12 + i * 3 + t];
        float qq = gelu_exact(acc + xs[ol][d + 1]);
        qbf[((size_t)(b * ED) + o) * DD + d] = (__bf16)qq;
        atomicAdd(&s1[ol], qq);
        atomicAdd(&s2[ol], qq * qq);
    }
    __syncthreads();
    if (d < 4) {
        atomicAdd(&sum[g * 4 + d],   s1[d]);
        atomicAdd(&sumsq[g * 4 + d], s2[d]);
    }
}

// ---------------- K2: BN affine finalize ----------------
__global__ void bn_finalize_kernel(const float* __restrict__ sum,
                                   const float* __restrict__ sumsq,
                                   const float* __restrict__ gamma,
                                   const float* __restrict__ beta,
                                   float* __restrict__ aArr, float* __restrict__ cArr)
{
    int o = threadIdx.x;                 // 512 threads
    const float n = (float)(NB * DD);
    float mu  = sum[o] / n;
    float var = sumsq[o] / n - mu * mu;
    float ae  = gamma[o] * rsqrtf(var + BN_EPS);
    aArr[o] = ae;
    cArr[o] = beta[o] - mu * ae;
}

// ---------------- K3: softmax over E axis + GELU -> k (bf16) ----------------
// grid 128 (batch), block 512: tid = d + 128*slice, slice covers 128 e's
__global__ __launch_bounds__(512) void ksoftmax_kernel(
        const float* __restrict__ x, __bf16* __restrict__ kbf)
{
    const int b = blockIdx.x;
    const int tid = threadIdx.x;
    const int d = tid & 127;
    const int s = tid >> 7;
    __shared__ float red[512];
    __shared__ float rowm[128];
    __shared__ float rows[128];
    const float* xb = x + (size_t)b * ED * DD;
    const int e0 = s * 128, e1 = e0 + 128;
    float m = -3.4e38f;
    for (int e = e0; e < e1; ++e) m = fmaxf(m, xb[e * DD + d]);
    red[tid] = m;
    __syncthreads();
    if (s == 0)
        rowm[d] = fmaxf(fmaxf(red[d], red[128 + d]), fmaxf(red[256 + d], red[384 + d]));
    __syncthreads();
    m = rowm[d];
    float sm = 0.f;
    for (int e = e0; e < e1; ++e) sm += __expf(xb[e * DD + d] - m);
    red[tid] = sm;
    __syncthreads();
    if (s == 0)
        rows[d] = red[d] + red[128 + d] + red[256 + d] + red[384 + d];
    __syncthreads();
    float inv = 1.0f / rows[d];
    for (int e = e0; e < e1; ++e) {
        float p = __expf(xb[e * DD + d] - m) * inv;
        kbf[((size_t)(b * ED) + e) * DD + d] = (__bf16)gelu_exact(p);
    }
}

// ---------------- K3b: ksum[b,f] = sum_d k[b,f,d] ----------------
// grid 8192, block 256 (8 waves, 1 row each)
__global__ __launch_bounds__(256) void ksum_kernel(const __bf16* __restrict__ kbf,
                                                   float* __restrict__ ksum)
{
    int row  = blockIdx.x * 8 + (threadIdx.x >> 5);
    int lane = threadIdx.x & 31;
    const __bf16* p = kbf + (size_t)row * DD;
    float s = 0.f;
    #pragma unroll
    for (int i = 0; i < 4; ++i) s += (float)p[lane + i * 32];
    #pragma unroll
    for (int off = 16; off >= 1; off >>= 1) s += __shfl_xor(s, off, 32);
    if (lane == 0) ksum[row] = s;
}

// ---------------- K4: v = gelu(x @ lin_w^T + b), stored transposed ----------------
// grid 4096 (= 128 batches * 32 M-tiles), block 256: wave w -> N-tile w*16
__global__ __launch_bounds__(256) void vgemm_kernel(
        const float* __restrict__ x, const float* __restrict__ lin_w,
        const float* __restrict__ lin_b, __bf16* __restrict__ vT)
{
    const int b  = blockIdx.x >> 5;
    const int Mt = blockIdx.x & 31;
    const int tid = threadIdx.x;
    const int w = tid >> 5, lane = tid & 31, lr = lane & 15;
    const int h8  = (lane & 16) ? 8 : 0;     // accumulator row offset per lane half
    const int h16 = (lane & 16) ? 16 : 0;    // B-frag K offset per lane half
    const int M0 = Mt * 16, N0 = w * 16;
    const float* arow = x + ((size_t)(b * ED) + M0 + lr) * DD;
    const float* brow = lin_w + (size_t)(N0 + lr) * DD;
    v8f acc = {};
    #pragma unroll
    for (int kk = 0; kk < 4; ++kk) {
        v16bf af, bf_;
        const float* ap = arow + kk * 32 + h16;
        #pragma unroll
        for (int j = 0; j < 8; ++j) { af[j] = (__bf16)ap[j]; af[8 + j] = (__bf16)ap[8 + j]; }
        const float* bp = brow + kk * 32 + h16;
        #pragma unroll
        for (int j = 0; j < 16; ++j) bf_[j] = (__bf16)bp[j];
        acc = wmma_bf16(af, bf_, acc);
    }
    int o = N0 + lr;
    float bias = lin_b[o];
    union { __bf16 h[8]; uint4 v; } pk;
    #pragma unroll
    for (int r = 0; r < 8; ++r) pk.h[r] = (__bf16)gelu_exact(acc[r] + bias);
    // 8 consecutive e's per lane -> one 16B store (16B-aligned: M0%16==0, h8 in {0,8})
    *(uint4*)(vT + ((size_t)(b * DD) + o) * ED + M0 + h8) = pk.v;
}

// ---------------- K5: fused logits + softmax + attn write + P*V^T ----------------
// grid (32 M-tiles, 128 batches), block 256 (8 waves)
__global__ __launch_bounds__(256) void attn_kernel(
        const __bf16* __restrict__ qbf, const __bf16* __restrict__ kbf,
        const __bf16* __restrict__ vT, const float* __restrict__ aArr,
        const float* __restrict__ cArr, const float* __restrict__ ksum,
        float* __restrict__ attn, float* __restrict__ out)
{
    const int Mt = blockIdx.x, b = blockIdx.y;
    const int R0 = Mt * 16;
    const int tid = threadIdx.x;
    const int w = tid >> 5, lane = tid & 31, lr = lane & 15;
    const int h8  = (lane & 16) ? 8 : 0;
    const int h16 = (lane & 16) ? 16 : 0;
    __shared__ float sS[16][520];            // 16 rows x 512 cols (+pad), 33 KB

    // stage 1: S' tile, BN folded in: (a_e * QK^T + c_e * ksum_f) * scale
    const __bf16* qrow = qbf + ((size_t)(b * ED) + R0 + lr) * DD;
    v8f acc[4];
    #pragma unroll
    for (int nt = 0; nt < 4; ++nt) { v8f z = {}; acc[nt] = z; }
    #pragma unroll
    for (int kk = 0; kk < 4; ++kk) {
        v16bf af;
        const __bf16* ap = qrow + kk * 32 + h16;
        #pragma unroll
        for (int j = 0; j < 8; ++j) { af[j] = ap[j]; af[8 + j] = ap[8 + j]; }
        #pragma unroll
        for (int nt = 0; nt < 4; ++nt) {
            int f = w * 64 + nt * 16 + lr;
            const __bf16* bp = kbf + ((size_t)(b * ED) + f) * DD + kk * 32 + h16;
            v16bf bf_;
            #pragma unroll
            for (int j = 0; j < 16; ++j) bf_[j] = bp[j];
            acc[nt] = wmma_bf16(af, bf_, acc[nt]);
        }
    }
    float ae[8], ce[8];
    #pragma unroll
    for (int r = 0; r < 8; ++r) { int e = R0 + r + h8; ae[r] = aArr[e]; ce[r] = cArr[e]; }
    #pragma unroll
    for (int nt = 0; nt < 4; ++nt) {
        int col = w * 64 + nt * 16 + lr;
        float ks = ksum[b * ED + col];
        #pragma unroll
        for (int r = 0; r < 8; ++r)
            sS[r + h8][col] = (ae[r] * acc[nt][r] + ce[r] * ks) * ATT_SCALE;
    }
    __syncthreads();

    // stage 2: row softmax (wave w owns rows 2w, 2w+1); write attn; P -> LDS
    #pragma unroll
    for (int rr = 0; rr < 2; ++rr) {
        int row = w * 2 + rr;
        float m = -3.4e38f;
        #pragma unroll
        for (int i = 0; i < 16; ++i) m = fmaxf(m, sS[row][lane + i * 32]);
        #pragma unroll
        for (int off = 16; off >= 1; off >>= 1) m = fmaxf(m, __shfl_xor(m, off, 32));
        float pv[16]; float sm = 0.f;
        #pragma unroll
        for (int i = 0; i < 16; ++i) { pv[i] = __expf(sS[row][lane + i * 32] - m); sm += pv[i]; }
        #pragma unroll
        for (int off = 16; off >= 1; off >>= 1) sm += __shfl_xor(sm, off, 32);
        float inv = 1.0f / sm;
        size_t arow = ((size_t)(b * ED) + R0 + row) * ED;
        #pragma unroll
        for (int i = 0; i < 16; ++i) {
            int col = lane + i * 32;
            float p = pv[i] * inv;
            attn[arow + col] = p;     // output #2 (f32), written once
            sS[row][col] = p;
        }
    }
    __syncthreads();

    // stage 3: out tile (16 x 128) = P (16x512, LDS) * vT^T (NT), K = 512
    const int N0 = w * 16;
    const __bf16* vrow = vT + ((size_t)(b * DD) + N0 + lr) * ED;
    v8f oacc = {};
    #pragma unroll
    for (int kk = 0; kk < 16; ++kk) {
        v16bf af;
        int ka = kk * 32 + h16;
        #pragma unroll
        for (int j = 0; j < 8; ++j) {
            af[j]     = (__bf16)sS[lr][ka + j];
            af[8 + j] = (__bf16)sS[lr][ka + 8 + j];
        }
        const __bf16* bp = vrow + kk * 32 + h16;
        v16bf bf_;
        #pragma unroll
        for (int j = 0; j < 16; ++j) bf_[j] = bp[j];
        oacc = wmma_bf16(af, bf_, oacc);
    }
    #pragma unroll
    for (int r = 0; r < 8; ++r) {
        int e = R0 + r + h8;
        out[((size_t)(b * ED) + e) * DD + N0 + lr] = oacc[r];
    }
}

extern "C" void kernel_launch(void* const* d_in, const int* in_sizes, int n_in,
                              void* d_out, int out_size, void* d_ws, size_t ws_size,
                              hipStream_t stream) {
    const float* x      = (const float*)d_in[0];
    const float* conv_w = (const float*)d_in[1];
    const float* conv_b = (const float*)d_in[2];
    const float* gamma  = (const float*)d_in[3];
    const float* beta   = (const float*)d_in[4];
    const float* lin_w  = (const float*)d_in[5];
    const float* lin_b  = (const float*)d_in[6];

    float* out  = (float*)d_out;                       // (128,512,128) f32
    float* attn = out + (size_t)NB * ED * DD;          // (128,512,512) f32

    char* ws = (char*)d_ws;
    size_t off = 0;
    auto suballoc = [&](size_t bytes) -> char* {
        char* p = ws + off;
        off = (off + bytes + 255) & ~(size_t)255;
        return p;
    };
    __bf16* qbf = (__bf16*)suballoc((size_t)NB * ED * DD * 2);   // 16 MB
    __bf16* kbf = (__bf16*)suballoc((size_t)NB * ED * DD * 2);   // 16 MB
    __bf16* vT  = (__bf16*)suballoc((size_t)NB * DD * ED * 2);   // 16 MB
    float* stats = (float*)suballoc(1024 * sizeof(float));        // sum|sumsq
    float* sum   = stats;
    float* sumsq = stats + 512;
    float* aArr  = (float*)suballoc(512 * sizeof(float));
    float* cArr  = (float*)suballoc(512 * sizeof(float));
    float* ksum  = (float*)suballoc((size_t)NB * ED * sizeof(float));

    init_zero_kernel<<<4, 256, 0, stream>>>(stats, 1024);
    conv_gelu_stats_kernel<<<dim3(128, 128), 128, 0, stream>>>(
        x, conv_w, conv_b, qbf, sum, sumsq);
    bn_finalize_kernel<<<1, 512, 0, stream>>>(sum, sumsq, gamma, beta, aArr, cArr);
    ksoftmax_kernel<<<NB, 512, 0, stream>>>(x, kbf);
    ksum_kernel<<<(NB * ED) / 8, 256, 0, stream>>>(kbf, ksum);
    vgemm_kernel<<<NB * 32, 256, 0, stream>>>(x, lin_w, lin_b, vT);
    attn_kernel<<<dim3(32, NB), 256, 0, stream>>>(
        qbf, kbf, vT, aArr, cArr, ksum, attn, out);
}
